// MoEFFN_5188320494402
// MI455X (gfx1250) — matrix operand order repaired and checked by reading, not compiled, not run on hardware
//
#include <hip/hip_runtime.h>
#include <hip/hip_bf16.h>
#include <math.h>

#define C_DIM   1024
#define H_DIM   4096
#define E_NUM   8
#define TOKENS  8192
#define ASSIGN  (TOKENS * 2)
#define TILE_M  16
#define MAX_TILES (ASSIGN / TILE_M)   // 1024 (worst case: one expert gets everything)

typedef unsigned short u16;
typedef __attribute__((ext_vector_type(16))) __bf16 v16bf;
typedef __attribute__((ext_vector_type(8)))  float  v8f;

union V16U { v16bf v; uint4 q[2]; };

__device__ __forceinline__ u16 f2bf(float f) {
    unsigned u = __float_as_uint(f);
    u += 0x7FFFu + ((u >> 16) & 1u);       // round-to-nearest-even
    return (u16)(u >> 16);
}

// CDNA5 async copy: global -> LDS, 16B per lane, tracked by ASYNCcnt.
__device__ __forceinline__ void async_copy_b128(void* lds_dst, const void* gsrc) {
    unsigned ldsoff = (unsigned)(uintptr_t)lds_dst;          // low 32 bits = LDS offset
    unsigned long long ga = (unsigned long long)(uintptr_t)gsrc;
    asm volatile("global_load_async_to_lds_b128 %0, %1, off"
                 :: "v"(ldsoff), "v"(ga)
                 : "memory");
}
__device__ __forceinline__ void wait_asynccnt0() {
    asm volatile("s_wait_asynccnt 0x0" ::: "memory");
}

// ---------------- zeroing ----------------
__global__ void k_zero_out(float4* __restrict__ out, int n4) {
    int i = blockIdx.x * blockDim.x + threadIdx.x;
    if (i < n4) out[i] = make_float4(0.f, 0.f, 0.f, 0.f);
}

__global__ void k_zero_meta(int* __restrict__ counts, int* __restrict__ cursors) {
    int i = threadIdx.x;
    if (i < E_NUM) { counts[i] = 0; cursors[i] = 0; }
}

// ---------------- fp32 -> bf16 (x, same layout) ----------------
__global__ void k_cvt_x(const float4* __restrict__ x, u16* __restrict__ xb, int n4) {
    int i = blockIdx.x * blockDim.x + threadIdx.x;
    if (i >= n4) return;
    float4 v = x[i];
    uint2 o;
    o.x = (unsigned)f2bf(v.x) | ((unsigned)f2bf(v.y) << 16);
    o.y = (unsigned)f2bf(v.z) | ((unsigned)f2bf(v.w) << 16);
    *(uint2*)(xb + (size_t)i * 4) = o;
}

// ------- tiled transpose + convert: src fp32 [e][R][Cc] -> dst bf16 [e][Cc][R] -------
__global__ void k_tcvt(const float* __restrict__ src, u16* __restrict__ dst, int R, int Cc) {
    __shared__ float tile[32][33];
    int e = blockIdx.z;
    const float* s = src + (size_t)e * R * Cc;
    u16*         d = dst + (size_t)e * R * Cc;
    int x  = blockIdx.x * 32 + threadIdx.x;   // col in src
    int y0 = blockIdx.y * 32;                 // row base in src
    for (int j = threadIdx.y; j < 32; j += 8)
        tile[j][threadIdx.x] = s[(size_t)(y0 + j) * Cc + x];
    __syncthreads();
    int xo  = y0 + threadIdx.x;               // col in dst (= src row)
    int yo0 = blockIdx.x * 32;                // row base in dst (= src col)
    for (int j = threadIdx.y; j < 32; j += 8)
        d[(size_t)(yo0 + j) * R + xo] = f2bf(tile[threadIdx.x][j]);
}

// ---------------- gating: softmax -> top2 -> renorm softmax, counts ----------------
__global__ void k_gate(const float* __restrict__ x, const float* __restrict__ Wg,
                       const float* __restrict__ bg, int* __restrict__ topk_idx,
                       float* __restrict__ topk_prob, int* __restrict__ counts) {
    int wave = threadIdx.x >> 5, lane = threadIdx.x & 31;
    int t = blockIdx.x * 8 + wave;            // one wave32 per token
    const float* xr = x + (size_t)t * C_DIM;
    float acc[E_NUM];
    #pragma unroll
    for (int e = 0; e < E_NUM; ++e) acc[e] = 0.f;
    for (int c = lane; c < C_DIM; c += 32) {
        float xv = xr[c];
        const float4* wr = (const float4*)(Wg + (size_t)c * E_NUM);
        float4 a = wr[0], b = wr[1];
        acc[0] += xv * a.x; acc[1] += xv * a.y; acc[2] += xv * a.z; acc[3] += xv * a.w;
        acc[4] += xv * b.x; acc[5] += xv * b.y; acc[6] += xv * b.z; acc[7] += xv * b.w;
    }
    #pragma unroll
    for (int e = 0; e < E_NUM; ++e)
        #pragma unroll
        for (int off = 16; off > 0; off >>= 1)
            acc[e] += __shfl_xor(acc[e], off, 32);
    // softmax over 8 experts (all lanes redundantly)
    float mx = -1e30f;
    #pragma unroll
    for (int e = 0; e < E_NUM; ++e) { acc[e] += bg[e]; mx = fmaxf(mx, acc[e]); }
    float sum = 0.f;
    #pragma unroll
    for (int e = 0; e < E_NUM; ++e) { acc[e] = expf(acc[e] - mx); sum += acc[e]; }
    float inv = 1.f / sum;
    #pragma unroll
    for (int e = 0; e < E_NUM; ++e) acc[e] *= inv;
    // top-2 (first index wins ties, like jax top_k)
    int i0 = 0; float v0 = acc[0];
    #pragma unroll
    for (int e = 1; e < E_NUM; ++e) if (acc[e] > v0) { v0 = acc[e]; i0 = e; }
    int i1 = (i0 == 0) ? 1 : 0; float v1 = acc[i1];
    #pragma unroll
    for (int e = 0; e < E_NUM; ++e) if (e != i0 && acc[e] > v1) { v1 = acc[e]; i1 = e; }
    // renormalized softmax over the two selected *probabilities* (matches reference)
    float e1 = expf(v1 - v0);
    float w0 = 1.f / (1.f + e1);
    float w1 = e1 * w0;
    if (lane == 0) {
        topk_idx[2 * t]     = i0;  topk_prob[2 * t]     = w0;
        topk_idx[2 * t + 1] = i1;  topk_prob[2 * t + 1] = w1;
        atomicAdd(&counts[i0], 1);
        atomicAdd(&counts[i1], 1);
    }
}

__global__ void k_scan(const int* __restrict__ counts, int* __restrict__ offsets) {
    if (threadIdx.x == 0 && blockIdx.x == 0) {
        int off = 0;
        for (int e = 0; e < E_NUM; ++e) { offsets[e] = off; off += counts[e]; }
    }
}

__global__ void k_scatter(const int* __restrict__ topk_idx, const float* __restrict__ topk_prob,
                          const int* __restrict__ offsets, int* __restrict__ cursors,
                          int* __restrict__ tok_list, float* __restrict__ prob_list) {
    int i = blockIdx.x * blockDim.x + threadIdx.x;
    if (i >= ASSIGN) return;
    int e = topk_idx[i];
    int pos = atomicAdd(&cursors[e], 1);
    int slot = offsets[e] + pos;
    tok_list[slot]  = i >> 1;     // token id
    prob_list[slot] = topk_prob[i];
}

// ---------------- fused grouped FFN: one 16-token tile per workgroup ----------------
#define XS_STRIDE (C_DIM + 8)     // bf16 elements, pad to dodge LDS bank aliasing
#define HS_STRIDE (H_DIM + 8)

#define WMMA_BF16(A, B, Cacc) \
    __builtin_amdgcn_wmma_f32_16x16x32_bf16(false, (A), false, (B), (short)0, (Cacc), false, false)

__global__ void __launch_bounds__(256)
k_moe(const u16* __restrict__ xb, const u16* __restrict__ w1t, const u16* __restrict__ w2t,
      const float* __restrict__ b1, const float* __restrict__ b2,
      const int* __restrict__ tok_list, const float* __restrict__ prob_list,
      const int* __restrict__ counts, const int* __restrict__ offsets,
      float* __restrict__ out) {
    extern __shared__ char smem[];
    u16*   Xs   = (u16*)smem;                                          // 16 x 1032 bf16
    u16*   Hs   = (u16*)(smem + (size_t)16 * XS_STRIDE * 2);           // 16 x 4104 bf16
    int*   toks = (int*)(smem + (size_t)16 * XS_STRIDE * 2 + (size_t)16 * HS_STRIDE * 2);
    float* prb  = (float*)(toks + TILE_M);

    int e    = blockIdx.y;
    int cnt  = counts[e];
    int row0 = blockIdx.x * TILE_M;
    if (row0 >= cnt) return;                 // empty tile for this expert
    int base = offsets[e];

    int tid = threadIdx.x;
    if (tid < TILE_M) {
        int r = min(row0 + tid, cnt - 1);    // clamp => duplicate rows, never stored
        toks[tid] = tok_list[base + r];
        prb[tid]  = prob_list[base + r];
    }
    __syncthreads();

    // gather X tile (16 rows x 1024 bf16) into LDS via CDNA5 async-to-LDS, 16B/lane
    #pragma unroll
    for (int i = 0; i < 8; ++i) {
        int g  = tid + i * 256;              // 2048 groups of 8 bf16
        int r  = g >> 7;
        int c8 = g & 127;
        const uint4* src = (const uint4*)(xb + (size_t)toks[r] * C_DIM) + c8;
        uint4*       dst = (uint4*)(Xs + (size_t)r * XS_STRIDE) + c8;
        async_copy_b128(dst, src);
    }
    wait_asynccnt0();
    __syncthreads();

    int wave   = tid >> 5, lane = tid & 31;
    int lm     = lane & 15;
    int khalfA = (lane < 16) ? 0 : 8;        // C-matrix M base (ISA 7.12.2); A K base is 2*khalfA (0 or 16)
    int khalfB = (lane < 16) ? 0 : 16;       // B-matrix K base

    const v8f Z = {0.f, 0.f, 0.f, 0.f, 0.f, 0.f, 0.f, 0.f};
    const u16* arowX = Xs + (size_t)lm * XS_STRIDE + khalfA;

    // ---- GEMM1: H = relu(X * W1 + b1), 16 x 4096, K = 1024 ----
    // 32 N-tiles per wave, processed 4 at a time: 1 A load feeds 4 WMMAs.
    for (int g = 0; g < 8; ++g) {
        int n0 = (wave + (g * 4 + 0) * 8) * 16;
        int n1 = (wave + (g * 4 + 1) * 8) * 16;
        int n2 = (wave + (g * 4 + 2) * 8) * 16;
        int n3 = (wave + (g * 4 + 3) * 8) * 16;
        v8f a0 = Z, a1 = Z, a2 = Z, a3 = Z;
        const u16* bp0 = w1t + ((size_t)e * H_DIM + n0 + lm) * C_DIM + khalfB;
        const u16* bp1 = w1t + ((size_t)e * H_DIM + n1 + lm) * C_DIM + khalfB;
        const u16* bp2 = w1t + ((size_t)e * H_DIM + n2 + lm) * C_DIM + khalfB;
        const u16* bp3 = w1t + ((size_t)e * H_DIM + n3 + lm) * C_DIM + khalfB;
        #pragma unroll 2
        for (int ks = 0; ks < 32; ++ks) {
            int k0 = ks * 32;
            V16U ua;
            ua.q[0] = *(const uint4*)(arowX + k0 + khalfA);
            ua.q[1] = *(const uint4*)(arowX + k0 + khalfA + 8);
            V16U u0, u1, u2, u3;
            u0.q[0] = *(const uint4*)(bp0 + k0); u0.q[1] = *(const uint4*)(bp0 + k0 + 8);
            u1.q[0] = *(const uint4*)(bp1 + k0); u1.q[1] = *(const uint4*)(bp1 + k0 + 8);
            u2.q[0] = *(const uint4*)(bp2 + k0); u2.q[1] = *(const uint4*)(bp2 + k0 + 8);
            u3.q[0] = *(const uint4*)(bp3 + k0); u3.q[1] = *(const uint4*)(bp3 + k0 + 8);
            a0 = WMMA_BF16(ua.v, u0.v, a0);
            a1 = WMMA_BF16(ua.v, u1.v, a1);
            a2 = WMMA_BF16(ua.v, u2.v, a2);
            a3 = WMMA_BF16(ua.v, u3.v, a3);
        }
        int nn[4] = {n0, n1, n2, n3};
        v8f  aa[4] = {a0, a1, a2, a3};
        #pragma unroll
        for (int i = 0; i < 4; ++i) {
            float bv = b1[e * H_DIM + nn[i] + lm];
            u16* hcol = Hs + nn[i] + lm;
            #pragma unroll
            for (int r = 0; r < 8; ++r) {
                float h = aa[i][r] + bv;
                h = h > 0.f ? h : 0.f;
                hcol[(size_t)(r + khalfA) * HS_STRIDE] = f2bf(h);
            }
        }
    }
    __syncthreads();

    // rows this lane's C-fragment touches (M = r + khalfA), preloaded once
    int   tk[8]; float pp[8];
    #pragma unroll
    for (int r = 0; r < 8; ++r) { tk[r] = toks[r + khalfA]; pp[r] = prb[r + khalfA]; }

    const u16* arowH = Hs + (size_t)lm * HS_STRIDE + khalfA;

    // ---- GEMM2: Out = H * W2 + b2, 16 x 1024, K = 4096 ----
    // 8 N-tiles per wave, processed 4 at a time.
    for (int g = 0; g < 2; ++g) {
        int n0 = (wave + (g * 4 + 0) * 8) * 16;
        int n1 = (wave + (g * 4 + 1) * 8) * 16;
        int n2 = (wave + (g * 4 + 2) * 8) * 16;
        int n3 = (wave + (g * 4 + 3) * 8) * 16;
        v8f a0 = Z, a1 = Z, a2 = Z, a3 = Z;
        const u16* bp0 = w2t + ((size_t)e * C_DIM + n0 + lm) * H_DIM + khalfB;
        const u16* bp1 = w2t + ((size_t)e * C_DIM + n1 + lm) * H_DIM + khalfB;
        const u16* bp2 = w2t + ((size_t)e * C_DIM + n2 + lm) * H_DIM + khalfB;
        const u16* bp3 = w2t + ((size_t)e * C_DIM + n3 + lm) * H_DIM + khalfB;
        #pragma unroll 2
        for (int ks = 0; ks < 128; ++ks) {
            int k0 = ks * 32;
            V16U ua;
            ua.q[0] = *(const uint4*)(arowH + k0 + khalfA);
            ua.q[1] = *(const uint4*)(arowH + k0 + khalfA + 8);
            V16U u0, u1, u2, u3;
            u0.q[0] = *(const uint4*)(bp0 + k0); u0.q[1] = *(const uint4*)(bp0 + k0 + 8);
            u1.q[0] = *(const uint4*)(bp1 + k0); u1.q[1] = *(const uint4*)(bp1 + k0 + 8);
            u2.q[0] = *(const uint4*)(bp2 + k0); u2.q[1] = *(const uint4*)(bp2 + k0 + 8);
            u3.q[0] = *(const uint4*)(bp3 + k0); u3.q[1] = *(const uint4*)(bp3 + k0 + 8);
            a0 = WMMA_BF16(ua.v, u0.v, a0);
            a1 = WMMA_BF16(ua.v, u1.v, a1);
            a2 = WMMA_BF16(ua.v, u2.v, a2);
            a3 = WMMA_BF16(ua.v, u3.v, a3);
        }
        int nn[4] = {n0, n1, n2, n3};
        v8f  aa[4] = {a0, a1, a2, a3};
        #pragma unroll
        for (int i = 0; i < 4; ++i) {
            float bv = b2[e * C_DIM + nn[i] + lm];
            #pragma unroll
            for (int r = 0; r < 8; ++r) {
                int grow = row0 + r + khalfA;
                if (grow < cnt) {
                    float o = (aa[i][r] + bv) * pp[r];
                    atomicAdd(&out[(size_t)tk[r] * C_DIM + nn[i] + lm], o);
                }
            }
        }
    }
}

extern "C" void kernel_launch(void* const* d_in, const int* in_sizes, int n_in,
                              void* d_out, int out_size, void* d_ws, size_t ws_size,
                              hipStream_t stream) {
    (void)in_sizes; (void)n_in; (void)out_size; (void)ws_size;
    const float* x  = (const float*)d_in[0];
    const float* W1 = (const float*)d_in[1];
    const float* b1 = (const float*)d_in[2];
    const float* W2 = (const float*)d_in[3];
    const float* b2 = (const float*)d_in[4];
    const float* Wg = (const float*)d_in[5];
    const float* bg = (const float*)d_in[6];
    float* out = (float*)d_out;

    char* ws = (char*)d_ws;
    size_t off = 0;
    auto carve = [&](size_t bytes) -> void* {
        void* p = ws + off;
        off += (bytes + 255) & ~(size_t)255;
        return p;
    };
    u16*   xb        = (u16*)carve((size_t)TOKENS * C_DIM * 2);
    u16*   w1t       = (u16*)carve((size_t)E_NUM * H_DIM * C_DIM * 2);
    u16*   w2t       = (u16*)carve((size_t)E_NUM * C_DIM * H_DIM * 2);
    int*   tki       = (int*)carve((size_t)ASSIGN * 4);
    float* tkp       = (float*)carve((size_t)ASSIGN * 4);
    int*   counts    = (int*)carve(256);
    int*   offsets   = (int*)carve(256);
    int*   cursors   = (int*)carve(256);
    int*   tok_list  = (int*)carve((size_t)ASSIGN * 4);
    float* prob_list = (float*)carve((size_t)ASSIGN * 4);

    int n4 = TOKENS * C_DIM / 4;   // 2,097,152 float4 / uint2 groups

    k_zero_out <<<dim3(n4 / 256), dim3(256), 0, stream>>>((float4*)out, n4);
    k_zero_meta<<<dim3(1), dim3(64), 0, stream>>>(counts, cursors);
    k_cvt_x    <<<dim3(n4 / 256), dim3(256), 0, stream>>>((const float4*)x, xb, n4);
    k_tcvt     <<<dim3(H_DIM / 32, C_DIM / 32, E_NUM), dim3(32, 8), 0, stream>>>(W1, w1t, C_DIM, H_DIM);
    k_tcvt     <<<dim3(C_DIM / 32, H_DIM / 32, E_NUM), dim3(32, 8), 0, stream>>>(W2, w2t, H_DIM, C_DIM);
    k_gate     <<<dim3(TOKENS / 8), dim3(256), 0, stream>>>(x, Wg, bg, tki, tkp, counts);
    k_scan     <<<dim3(1), dim3(1), 0, stream>>>(counts, offsets);
    k_scatter  <<<dim3(ASSIGN / 256), dim3(256), 0, stream>>>(tki, tkp, offsets, cursors,
                                                              tok_list, prob_list);
    size_t lds = (size_t)16 * XS_STRIDE * 2 + (size_t)16 * HS_STRIDE * 2
               + (size_t)TILE_M * 4 + (size_t)TILE_M * 4;   // ~164.5 KB of the 320 KB WGP LDS
    k_moe      <<<dim3(MAX_TILES, E_NUM), dim3(256), lds, stream>>>(
                   xb, w1t, w2t, b1, b2, tok_list, prob_list, counts, offsets, out);
}
